// VectorQuantizer_472446402881
// MI455X (gfx1250) — matrix operand order repaired and checked by reading, not compiled, not run on hardware
//
#include <hip/hip_runtime.h>

typedef __attribute__((ext_vector_type(16))) _Float16 v16h;
typedef __attribute__((ext_vector_type(8)))  float    v8f;

#define D_DIM          64
#define K_CODES        512
#define N_TILES        32          // K_CODES / 16
#define WAVES_PER_BLK  8
#define ROWS_PER_BLK   128         // 8 waves * 16 rows

// workspace layout (bytes), all offsets 256B aligned
#define WS_BPACK  0                // 32 tiles * 2 steps * 32 lanes * 16 f16 = 65536 B
#define WS_EMBT   65536            // 512 * 64 fp32 = 131072 B
#define WS_ENORM  196608           // 512 fp32 = 2048 B
#define WS_PART   198656           // nBlocks fp32

// ---------------------------------------------------------------------------
// Precompute: pack embeddings into the WMMA B-matrix lane layout (f16),
// build transposed fp32 embeddings (for exact output gather) and ||e_k||^2.
// B packing used here (contiguous k, same as A; B is 32x16): lane = n + 16*h holds column n,
// element j of v16h is K = j + 16*h  (global d = step*32 + j + 16*h).
// ---------------------------------------------------------------------------
__global__ __launch_bounds__(256) void vq_pre_kernel(
    const float* __restrict__ emb,   // (64, 512) row-major
    _Float16* __restrict__ Bpack,    // [tile][step][lane][16]
    float* __restrict__ embT,        // (512, 64)
    float* __restrict__ enorm)       // (512)
{
    const int tid = blockIdx.x * 256 + threadIdx.x;   // 0..2047
    // ---- Bpack: one (tile, step, lane) fragment per thread ----
    {
        const int lane = tid & 31;
        const int s    = (tid >> 5) & 1;
        const int t    = tid >> 6;
        const int n    = t * 16 + (lane & 15);
        const int dbase = s * 32 + 16 * (lane >> 4);
        _Float16* bp = Bpack + (size_t)tid * 16;
        #pragma unroll
        for (int j = 0; j < 16; ++j)
            bp[j] = (_Float16)emb[(size_t)(dbase + j) * K_CODES + n];
    }
    // ---- embT: 16 contiguous elements per thread ----
    {
        #pragma unroll
        for (int u = 0; u < 16; ++u) {
            int f = tid * 16 + u;          // 0..32767
            int k = f >> 6;                // code index
            int d = f & 63;                // dim index
            embT[f] = emb[(size_t)d * K_CODES + k];
        }
    }
    // ---- enorm ----
    if (tid < K_CODES) {
        float s2 = 0.f;
        #pragma unroll
        for (int d = 0; d < D_DIM; ++d) {
            float v = emb[(size_t)d * K_CODES + tid];
            s2 += v * v;
        }
        enorm[tid] = s2;
    }
}

// ---------------------------------------------------------------------------
// Main kernel: each wave32 handles 16 rows of x against all 512 codes.
// sim tile computed with 4 chained v_wmma_f32_16x16x32_f16 (x split hi+lo).
// ---------------------------------------------------------------------------
__global__ __launch_bounds__(256) void vq_main_kernel(
    const float* __restrict__ x,        // (N, 64)
    const _Float16* __restrict__ Bpack,
    const float* __restrict__ embT,     // (512, 64)
    const float* __restrict__ enorm,    // (512)
    float* __restrict__ out,            // (N, 64)
    float* __restrict__ partials)       // (gridDim.x)
{
    const int lane = threadIdx.x & 31;
    const int wave = threadIdx.x >> 5;
    const int h    = lane >> 4;         // lane half
    const int m    = lane & 15;         // row within tile (A layout)
    const long rowBase = ((long)blockIdx.x * WAVES_PER_BLK + wave) * 16;

    // --- load A fragments: contiguous k order used here (same as Bpack): K = j + 16h
    const float* xr = x + (rowBase + m) * D_DIM;
    v16h a0h, a0l, a1h, a1l;
    #pragma unroll
    for (int j = 0; j < 16; ++j) {
        const int K = j + 16 * h;
        float v0 = xr[K];
        float v1 = xr[32 + K];
        _Float16 h0 = (_Float16)v0;
        _Float16 h1 = (_Float16)v1;
        a0h[j] = h0;  a0l[j] = (_Float16)(v0 - (float)h0);
        a1h[j] = h1;  a1l[j] = (_Float16)(v1 - (float)h1);
    }

    float best[8];
    int   bidx[8];
    #pragma unroll
    for (int r = 0; r < 8; ++r) { best[r] = 3.0e38f; bidx[r] = 0x7fffffff; }

    const v16h* bp = (const v16h*)Bpack;
    const int ncol = (lane & 15);       // code column within tile (C/D layout)

    for (int t = 0; t < N_TILES; ++t) {
        v16h b0 = bp[(t * 2 + 0) * 32 + lane];
        v16h b1 = bp[(t * 2 + 1) * 32 + lane];
        v8f c = {};
        c = __builtin_amdgcn_wmma_f32_16x16x32_f16(false, a0h, false, b0,
                                                   (short)0, c, false, false);
        c = __builtin_amdgcn_wmma_f32_16x16x32_f16(false, a0l, false, b0,
                                                   (short)0, c, false, false);
        c = __builtin_amdgcn_wmma_f32_16x16x32_f16(false, a1h, false, b1,
                                                   (short)0, c, false, false);
        c = __builtin_amdgcn_wmma_f32_16x16x32_f16(false, a1l, false, b1,
                                                   (short)0, c, false, false);
        const int n  = (t << 4) + ncol;
        const float en = enorm[n];
        #pragma unroll
        for (int r = 0; r < 8; ++r) {
            // dist = ||e||^2 - 2*sim  (||x||^2 per-row constant: same argmin)
            float dist = __builtin_fmaf(-2.0f, c[r], en);
            if (dist < best[r]) { best[r] = dist; bidx[r] = n; }  // first-min
        }
    }

    // --- cross-lane argmin over the 16 lanes of each half (rows M=r+8h) ---
    #pragma unroll
    for (int r = 0; r < 8; ++r) {
        #pragma unroll
        for (int off = 1; off < 16; off <<= 1) {
            float ov = __shfl_xor(best[r], off, 32);
            int   oi = __shfl_xor(bidx[r], off, 32);
            if (ov < best[r] || (ov == best[r] && oi < bidx[r])) {
                best[r] = ov; bidx[r] = oi;
            }
        }
    }

    // --- gather exact fp32 codewords, write output, accumulate sq error ---
    float errAcc = 0.f;
    #pragma unroll
    for (int mm = 0; mm < 16; ++mm) {
        const int idx = __shfl(bidx[mm & 7], (mm >> 3) << 4, 32);
        const long base = (rowBase + mm) * D_DIM;
        const float* eq = embT + (long)idx * D_DIM;
        float q0 = eq[lane];
        float q1 = eq[lane + 32];
        float xv0 = x[base + lane];
        float xv1 = x[base + lane + 32];
        out[base + lane]      = q0;
        out[base + lane + 32] = q1;
        float e0 = q0 - xv0;
        float e1 = q1 - xv1;
        errAcc += e0 * e0 + e1 * e1;
    }

    // --- deterministic block reduction of squared error ---
    #pragma unroll
    for (int off = 16; off >= 1; off >>= 1)
        errAcc += __shfl_xor(errAcc, off, 32);
    __shared__ float sw[WAVES_PER_BLK];
    if (lane == 0) sw[wave] = errAcc;
    __syncthreads();
    if (threadIdx.x == 0) {
        float s = 0.f;
        #pragma unroll
        for (int i = 0; i < WAVES_PER_BLK; ++i) s += sw[i];
        partials[blockIdx.x] = s;
    }
}

// ---------------------------------------------------------------------------
// Final reduction: loss = 1.25 * sum(sq err) / totalElems  (fixed order)
// ---------------------------------------------------------------------------
__global__ __launch_bounds__(256) void vq_reduce_kernel(
    const float* __restrict__ partials, float* __restrict__ loss,
    int n, float totalElems)
{
    __shared__ float s[256];
    float a = 0.f;
    for (int i = threadIdx.x; i < n; i += 256) a += partials[i];
    s[threadIdx.x] = a;
    __syncthreads();
    for (int k = 128; k > 0; k >>= 1) {
        if (threadIdx.x < k) s[threadIdx.x] += s[threadIdx.x + k];
        __syncthreads();
    }
    if (threadIdx.x == 0) *loss = s[0] * 1.25f / totalElems;
}

extern "C" void kernel_launch(void* const* d_in, const int* in_sizes, int n_in,
                              void* d_out, int out_size, void* d_ws, size_t ws_size,
                              hipStream_t stream) {
    const float* x   = (const float*)d_in[0];   // (64,64,64,64) fp32
    const float* emb = (const float*)d_in[1];   // (64,512) fp32
    float* out = (float*)d_out;                 // N*64 quantized + 1 loss

    char* ws = (char*)d_ws;
    _Float16* Bpack   = (_Float16*)(ws + WS_BPACK);
    float*    embT    = (float*)(ws + WS_EMBT);
    float*    enorm   = (float*)(ws + WS_ENORM);
    float*    partials= (float*)(ws + WS_PART);

    const int totalElems = in_sizes[0];          // 16,777,216
    const int nRows   = totalElems / D_DIM;      // 262,144
    const int nBlocks = nRows / ROWS_PER_BLK;    // 2,048

    vq_pre_kernel<<<8, 256, 0, stream>>>(emb, Bpack, embT, enorm);
    vq_main_kernel<<<nBlocks, 256, 0, stream>>>(x, Bpack, embT, enorm, out, partials);
    vq_reduce_kernel<<<1, 256, 0, stream>>>(partials, out + (size_t)totalElems,
                                            nBlocks, (float)totalElems);
}
